// DSConv_28381143892655
// MI455X (gfx1250) — matrix operand order repaired and checked by reading, not compiled, not run on hardware
//
#include <hip/hip_runtime.h>
#include <hip/hip_bf16.h>

typedef __attribute__((ext_vector_type(16))) _Float16 v16h;
typedef __attribute__((ext_vector_type(8)))  _Float16 v8h;
typedef __attribute__((ext_vector_type(8)))  float    v8f;
typedef __attribute__((ext_vector_type(4)))  uint32_t v4u;
typedef __attribute__((ext_vector_type(8)))  uint32_t v8u;

#define B_    8
#define CIN   64
#define COUT  64
#define H_    128
#define W_    128
#define HW_   (H_ * W_)          // 16384
#define KK9   9
#define KDIM  (CIN * KK9)        // 576
#define LK_   15
#define LP_   7
#define BPAD  16                 // LDS row padding (halves)

// ---------------------------------------------------------------------------
// Kernel 0: convert deform weights (64 x 576) f32 -> f16 once per launch.
// ---------------------------------------------------------------------------
__global__ __launch_bounds__(256) void wcvt_kernel(
    const float* __restrict__ wd, _Float16* __restrict__ wdh)
{
    int i = blockIdx.x * 256 + threadIdx.x;     // 36,864
    wdh[i] = (_Float16)wd[i];
}

// ---------------------------------------------------------------------------
// Kernel 1: offset = hconv(1x15) + vconv(15x1), mask = sigmoid(conv3x3)
// One thread per (b,h,w); all 27 output channels accumulated together so the
// 64*(15+15+9) input reads are reused 18x/9x. Weight indices are wave-uniform
// -> compiler scalarizes them into SMEM loads.
// ---------------------------------------------------------------------------
__global__ __launch_bounds__(256) void offmask_kernel(
    const float* __restrict__ x,
    const float* __restrict__ wh, const float* __restrict__ bh,
    const float* __restrict__ wv, const float* __restrict__ bv,
    const float* __restrict__ wm, const float* __restrict__ bm,
    float* __restrict__ off, float* __restrict__ msk)
{
    int idx = blockIdx.x * 256 + threadIdx.x;      // B*H*W = 131072
    int b  = idx >> 14;
    int hw = idx & (HW_ - 1);
    int h  = hw >> 7;
    int w  = hw & (W_ - 1);
    const float* xb = x + ((size_t)b << 20);       // b * 64 * 16384

    float acc[18];
    float mac[9];
#pragma unroll
    for (int i = 0; i < 18; ++i) acc[i] = 0.0f;
#pragma unroll
    for (int i = 0; i < 9; ++i)  mac[i] = 0.0f;

#pragma unroll 1
    for (int c = 0; c < CIN; ++c) {
        const float* xc = xb + (c << 14);
        float hrow[LK_], vcol[LK_], nb[9];
#pragma unroll
        for (int t = 0; t < LK_; ++t) {
            int ww = w - LP_ + t;
            hrow[t] = (ww >= 0 && ww < W_) ? xc[h * W_ + ww] : 0.0f;
        }
#pragma unroll
        for (int t = 0; t < LK_; ++t) {
            int hh = h - LP_ + t;
            vcol[t] = (hh >= 0 && hh < H_) ? xc[hh * W_ + w] : 0.0f;
        }
#pragma unroll
        for (int j = 0; j < 9; ++j) {
            int hh = h - 1 + (j / 3);
            int ww = w - 1 + (j % 3);
            nb[j] = (hh >= 0 && hh < H_ && ww >= 0 && ww < W_) ? xc[hh * W_ + ww] : 0.0f;
        }
#pragma unroll 1
        for (int co = 0; co < 18; ++co) {
            const float* whp = wh + (size_t)(co * CIN + c) * LK_;
            const float* wvp = wv + (size_t)(co * CIN + c) * LK_;
            float s = 0.0f;
#pragma unroll
            for (int t = 0; t < LK_; ++t)
                s = fmaf(hrow[t], whp[t], fmaf(vcol[t], wvp[t], s));
            acc[co] += s;
        }
#pragma unroll 1
        for (int co = 0; co < 9; ++co) {
            const float* wmp = wm + (size_t)(co * CIN + c) * 9;
            float s = 0.0f;
#pragma unroll
            for (int j = 0; j < 9; ++j)
                s = fmaf(nb[j], wmp[j], s);
            mac[co] += s;
        }
    }

#pragma unroll
    for (int co = 0; co < 18; ++co)
        off[(((size_t)b * 18 + co) << 14) + hw] = acc[co] + bh[co] + bv[co];
#pragma unroll
    for (int co = 0; co < 9; ++co) {
        float v = mac[co] + bm[co];
        msk[(((size_t)b * 9 + co) << 14) + hw] = 1.0f / (1.0f + expf(-v));
    }
}

// ---------------------------------------------------------------------------
// Kernel 2: deformable conv as gather + WMMA GEMM.
// Block = 32 consecutive spatial positions of one batch image (one row chunk).
//  - Wave 0 kicks off a Tensor Data Mover load of the full f16 weight matrix
//    (64 x 576, 72 KB) into LDS; it overlaps with the gather phase and is
//    fenced with s_wait_tensorcnt before the MMA phase.
//  - Phase 1: 256 threads gather/bilinear-sample the 576 x 32 B-matrix into
//    LDS (f16, layout Bsm[position][k] -> a lane's 16 K-halves are two contiguous runs of 8).
//  - Phase 2: 8 waves; wave w owns M-tile (w&3) and N-tile (w>>2); 18 steps of
//    v_wmma_f32_16x16x32_f16 over K=576, fragments loaded as ds_load_b128.
// ---------------------------------------------------------------------------
__global__ __launch_bounds__(256) void deform_wmma_kernel(
    const float*    __restrict__ x,
    const float*    __restrict__ off,
    const float*    __restrict__ msk,
    const _Float16* __restrict__ wdh,   // (64, 576) f16 row-major
    float*          __restrict__ y)
{
    __shared__ __attribute__((aligned(16))) _Float16 Asm[COUT][KDIM];       // 73,728 B
    __shared__ __attribute__((aligned(16))) _Float16 Bsm[32][KDIM + BPAD];  // 37,888 B

    int tile   = blockIdx.x;                 // B * HW / 32 = 4096
    int b      = tile >> 9;                  // HW/32 = 512 tiles per image
    int hwbase = (tile & 511) * 32;
    const float* xb = x + ((size_t)b << 20);

    // ---- Kick off TDM: global f16 weights -> LDS A-matrix (async) ----
    if (threadIdx.x < 32) {
        uint64_t ga      = (uint64_t)(uintptr_t)wdh;
        uint32_t lds_off = (uint32_t)(uintptr_t)&Asm[0][0];  // low 32 = LDS offset
        v4u g0;
        g0[0] = 1u;                                   // count=1, no gather/restore
        g0[1] = lds_off;                              // lds_addr [63:32]
        g0[2] = (uint32_t)ga;                         // global_addr [95:64]
        g0[3] = ((uint32_t)(ga >> 32) & 0x01FFFFFFu)  // global_addr [120:96]
              | 0x80000000u;                          // type=2 ("image") [127:126]
        v8u g1;
        g1[0] = 0x00010000u;          // wg_mask=0, data_size=1 (2B)
        g1[1] = (KDIM & 0xFFFFu) << 16;  // tensor_dim0[15:0] @ bits 63:48
        g1[2] = (COUT & 0xFFFFu) << 16;  // tensor_dim1[15:0] @ bits 95:80
        g1[3] = (KDIM & 0xFFFFu) << 16;  // tile_dim0 @ bits 127:112
        g1[4] = COUT;                    // tile_dim1 @ bits 143:128
        g1[5] = KDIM;                    // tensor_dim0_stride low 32
        g1[6] = 0u;
        g1[7] = 0u;
        asm volatile("tensor_load_to_lds %0, %1" :: "s"(g0), "s"(g1) : "memory");
    }

    // ---- Phase 1: build sampled B-matrix (overlaps with TDM) ----
    for (int pair = threadIdx.x; pair < 32 * KK9; pair += 256) {
        int n   = pair & 31;
        int tap = pair >> 5;                 // 0..8
        int hw  = hwbase + n;
        int h   = hw >> 7;
        int w   = hw & (W_ - 1);
        int ky  = tap / 3;
        int kx  = tap % 3;

        float dy = off[(((size_t)b * 18 + tap * 2 + 0) << 14) + hw];
        float dx = off[(((size_t)b * 18 + tap * 2 + 1) << 14) + hw];
        float mv = msk[(((size_t)b * 9 + tap) << 14) + hw];

        float py = dy + (float)(ky + h - 1);
        float px = dx + (float)(kx + w - 1);
        float y0f = floorf(py), x0f = floorf(px);
        int   y0  = (int)y0f,   x0  = (int)x0f;
        float wy1 = py - y0f, wy0 = 1.0f - wy1;
        float wx1 = px - x0f, wx0 = 1.0f - wx1;

        int   ys[2] = { y0, y0 + 1 };
        int   xs[2] = { x0, x0 + 1 };
        float wys[2] = { wy0, wy1 };
        float wxs[2] = { wx0, wx1 };

        float cw[4];
        int   ci[4];
#pragma unroll
        for (int q = 0; q < 4; ++q) {
            int yy = ys[q >> 1], xx = xs[q & 1];
            bool valid = (yy >= 0) & (yy < H_) & (xx >= 0) & (xx < W_);
            int yc = min(max(yy, 0), H_ - 1);
            int xc = min(max(xx, 0), W_ - 1);
            ci[q] = yc * W_ + xc;
            cw[q] = wys[q >> 1] * wxs[q & 1] * (valid ? 1.0f : 0.0f);
        }

#pragma unroll 4
        for (int c = 0; c < CIN; ++c) {
            const float* xc = xb + ((size_t)c << 14);
            float s = cw[0] * xc[ci[0]] + cw[1] * xc[ci[1]]
                    + cw[2] * xc[ci[2]] + cw[3] * xc[ci[3]];
            Bsm[n][c * KK9 + tap] = (_Float16)(s * mv);
        }
    }

    // ---- Fence the TDM transfer, then barrier so every wave sees A + B ----
    if (threadIdx.x < 32) __builtin_amdgcn_s_wait_tensorcnt(0);
    __syncthreads();

    // ---- Phase 2: WMMA ----
    int lane  = threadIdx.x & 31;
    int wave  = threadIdx.x >> 5;
    int mtile = wave & 3;                    // 4 M-tiles of 16 (Cout=64)
    int ntile = wave >> 2;                   // 2 N-tiles of 16 (32 positions)
    int ml    = lane & 15;
    int khA   = (lane < 16) ? 0 : 8;         // A: lanes 16-31 hold K+8
    int krB   = (lane < 16) ? 0 : 8;         // B: lanes 16-31 hold K+8
    int col   = ntile * 16 + ml;             // local position 0..31

    const _Float16* arow = &Asm[mtile * 16 + ml][0];
    const _Float16* brow = &Bsm[col][0];

    v8f acc = {};
#pragma unroll 2
    for (int kb = 0; kb < KDIM; kb += 32) {
        v8h a0 = *(const v8h*)(arow + kb + khA);        // K = khA .. khA+7
        v8h a1 = *(const v8h*)(arow + kb + khA + 16);   // K = khA+16 .. khA+23
        v8h b0 = *(const v8h*)(brow + kb + krB);        // K = krB .. krB+7
        v8h b1 = *(const v8h*)(brow + kb + krB + 16);   // K = krB+16 .. krB+23
        v16h afrag = __builtin_shufflevector(a0, a1, 0,1,2,3,4,5,6,7,8,9,10,11,12,13,14,15);
        v16h bfrag = __builtin_shufflevector(b0, b1, 0,1,2,3,4,5,6,7,8,9,10,11,12,13,14,15);
        acc = __builtin_amdgcn_wmma_f32_16x16x32_f16(
                  false, afrag, false, bfrag, (short)0, acc, false, false);
    }

    // ---- Store C tile: VGPR j -> M = j (+8 for lanes 16-31), N = lane&15 ----
    int hw0  = hwbase + col;
    int moff = (lane < 16) ? 0 : 8;
#pragma unroll
    for (int j = 0; j < 8; ++j) {
        int m = mtile * 16 + moff + j;
        y[(((size_t)b * COUT + m) << 14) + hw0] = acc[j];
    }
}

// ---------------------------------------------------------------------------
// Kernel 3: per-channel sum / sum-of-squares. One block per channel -> direct
// store, no atomics, no zero-init pass.
// ---------------------------------------------------------------------------
__global__ __launch_bounds__(256) void stats_kernel(
    const float* __restrict__ y, float* __restrict__ st)
{
    __shared__ float sS[256], sQ[256];
    int c   = blockIdx.x;
    int tid = threadIdx.x;
    float s = 0.0f, q = 0.0f;
    for (int e = tid; e < B_ * HW_; e += 256) {
        int b  = e >> 14;
        int hw = e & (HW_ - 1);
        float v = y[(((size_t)b * COUT + c) << 14) + hw];
        s += v;
        q += v * v;
    }
    sS[tid] = s; sQ[tid] = q;
    __syncthreads();
    for (int o = 128; o > 0; o >>= 1) {
        if (tid < o) { sS[tid] += sS[tid + o]; sQ[tid] += sQ[tid + o]; }
        __syncthreads();
    }
    if (tid == 0) { st[c] = sS[0]; st[COUT + c] = sQ[0]; }
}

// ---------------------------------------------------------------------------
// Kernel 4: BN normalize + affine + ReLU -> d_out
// ---------------------------------------------------------------------------
__global__ __launch_bounds__(256) void norm_kernel(
    const float* __restrict__ y, const float* __restrict__ st,
    const float* __restrict__ gamma, const float* __restrict__ beta,
    float* __restrict__ out)
{
    int i = blockIdx.x * 256 + threadIdx.x;    // 8,388,608 elements
    int c = (i >> 14) & (COUT - 1);
    const float invN = 1.0f / (float)(B_ * HW_);
    float mean = st[c] * invN;
    float var  = st[COUT + c] * invN - mean * mean;
    float v = (y[i] - mean) * rsqrtf(var + 1e-5f);
    v = v * gamma[c] + beta[c];
    out[i] = fmaxf(v, 0.0f);
}

// ---------------------------------------------------------------------------
extern "C" void kernel_launch(void* const* d_in, const int* in_sizes, int n_in,
                              void* d_out, int out_size, void* d_ws, size_t ws_size,
                              hipStream_t stream)
{
    const float* x     = (const float*)d_in[0];
    const float* wh    = (const float*)d_in[1];
    const float* bh    = (const float*)d_in[2];
    const float* wv    = (const float*)d_in[3];
    const float* bv    = (const float*)d_in[4];
    const float* wm    = (const float*)d_in[5];
    const float* bm    = (const float*)d_in[6];
    const float* wd    = (const float*)d_in[7];
    const float* gamma = (const float*)d_in[8];
    const float* beta  = (const float*)d_in[9];
    float* out = (float*)d_out;

    // workspace partition (floats): offset | mask | y | stats | f16 weights
    float* off = (float*)d_ws;                       // 8*18*16384 = 2,359,296
    float* msk = off + (size_t)B_ * 18 * HW_;        // 8* 9*16384 = 1,179,648
    float* y   = msk + (size_t)B_ * 9 * HW_;         // 8*64*16384 = 8,388,608
    float* st  = y   + (size_t)B_ * COUT * HW_;      // 128
    _Float16* wdh = (_Float16*)(st + 128);           // 36,864 halves

    wcvt_kernel<<<(COUT * KDIM) / 256, 256, 0, stream>>>(wd, wdh);

    offmask_kernel<<<(B_ * HW_) / 256, 256, 0, stream>>>(
        x, wh, bh, wv, bv, wm, bm, off, msk);

    deform_wmma_kernel<<<(B_ * HW_) / 32, 256, 0, stream>>>(
        x, off, msk, wdh, y);

    stats_kernel<<<COUT, 256, 0, stream>>>(y, st);

    norm_kernel<<<(B_ * COUT * HW_) / 256, 256, 0, stream>>>(
        y, st, gamma, beta, out);
}
